// MoELayer_5652176962260
// MI455X (gfx1250) — matrix operand order repaired and checked by reading, not compiled, not run on hardware
//
#include <hip/hip_runtime.h>
#include <hip/hip_bf16.h>
#include <math.h>

// MoE forward for MI455X (gfx1250, wave32, WMMA + Tensor Data Mover).
// d_in order: x, attention_mask(unused), gate_w, W1, b1, W2, b2  (all fp32)
// d_out: [T*D] out fp32, then [1] balance_loss, then [8] counts (as float).

#define T_TOK 8192
#define DMODEL 768
#define HDIM 3072
#define NEXP 8
#define MT 64            // tokens per workgroup tile (4 M-tiles of 16)
#define HC 64            // H chunk width
#define NCH (HDIM / HC)  // 48 chunks
#define NTHREADS 512     // 16 waves

typedef __attribute__((ext_vector_type(16))) __bf16 v16bf;
typedef __attribute__((ext_vector_type(8)))  float  v8f;

union FragU { uint4 u[2]; v16bf v; };

static __device__ __forceinline__ unsigned short f2bf(float f) {
  unsigned u = __builtin_bit_cast(unsigned, f);
  unsigned r = u + 0x7FFFu + ((u >> 16) & 1u);
  return (unsigned short)(r >> 16);
}

// A fragment: 16x32 bf16 tile, row-major LDS (stride elements/row).
// Lane l: M=l&15, K-base=(l>>4)*8; V0..3 = K kb..kb+7, V4..7 = K kb+16..kb+23.
static __device__ __forceinline__ v16bf load_frag_A(
    const unsigned short* lds, int row0, int k0, int stride, int lane) {
  int m = lane & 15, g = lane >> 4;
  const unsigned short* p = lds + (row0 + m) * stride + k0 + g * 8;
  FragU f;
  f.u[0] = *(const uint4*)(p);
  f.u[1] = *(const uint4*)(p + 16);
  return f.v;
}

// B fragment: 32x16 bf16 (KxN); LDS holds a contiguous K run per column
// (kstride elements). Lane l: N=l&15, K-base=k0+(l>>4)*8; V0..3 = K kb..kb+7, V4..7 = K kb+16..kb+23.
static __device__ __forceinline__ v16bf load_frag_B(
    const unsigned short* lds, int col0, int k0, int kstride, int lane) {
  int c = lane & 15, g = lane >> 4;
  const unsigned short* p = lds + (col0 + c) * kstride + k0 + g * 8;
  FragU f;
  f.u[0] = *(const uint4*)(p);
  f.u[1] = *(const uint4*)(p + 16);
  return f.v;
}

// ---------------- Tensor Data Mover helpers ----------------
#if __has_builtin(__builtin_amdgcn_tensor_load_to_lds) && \
    __has_builtin(__builtin_amdgcn_s_wait_tensorcnt)
#define USE_TDM 1
typedef unsigned int v4u __attribute__((ext_vector_type(4)));
typedef int v8i_t __attribute__((ext_vector_type(8)));
typedef int v4i_t __attribute__((ext_vector_type(4)));

// 6-arg builtin (clang-23 / therock signature):
// (uint32x4 g0, int32x8 g1, int32x4 g2, int32x4 g3, int32x8 pad, i32 cpol)
static __device__ __forceinline__ void tdm_issue(v4u g0, v8i_t g1, v4i_t g2,
                                                 v4i_t g3) {
  v8i_t zpad = {0, 0, 0, 0, 0, 0, 0, 0};
  __builtin_amdgcn_tensor_load_to_lds(g0, g1, g2, g3, zpad, 0);
}

// 2D tile load: tile_d1 rows of tile_d0 bf16 elements, row stride stride0
// (elements), packed contiguously at lds_off. D# fields per ISA 08_async_tensor.
static __device__ __forceinline__ void tdm_load_2d(
    unsigned lds_off, const void* gptr, unsigned tensor_d0, unsigned tensor_d1,
    unsigned tile_d0, unsigned tile_d1, unsigned stride0) {
  unsigned long long ga = (unsigned long long)gptr;
  v4u g0;
  g0.x = 1u;                                    // count=1, user descriptor
  g0.y = lds_off;                               // lds_addr (bytes)
  g0.z = (unsigned)ga;                          // global_addr lo
  g0.w = (unsigned)((ga >> 32) & 0x01FFFFFFu) | (2u << 30);  // addr hi | type=2
  v8i_t g1;
  g1[0] = 0x00010000;                           // data_size=1 (2 bytes)
  g1[1] = (int)((tensor_d0 & 0xFFFFu) << 16);
  g1[2] = (int)((tensor_d0 >> 16) | ((tensor_d1 & 0xFFFFu) << 16));
  g1[3] = (int)((tensor_d1 >> 16) | (tile_d0 << 16));
  g1[4] = (int)(tile_d1 & 0xFFFFu);             // tile_dim1 | tile_dim2=0
  g1[5] = (int)stride0;                         // tensor_dim0_stride lo
  g1[6] = 0;
  g1[7] = 0;
  v4i_t z = {0, 0, 0, 0};
  tdm_issue(g0, g1, z, z);
}

// Gather-mode load: up to 16 rows (16-bit indices) of tile_d0 bf16 elements
// from a [nrows x tensor_d0] tensor, packed at lds_off.
static __device__ __forceinline__ void tdm_gather_rows(
    unsigned lds_off, const void* base, const int* idx, int nvalid,
    unsigned tensor_d0, unsigned nrows) {
  unsigned long long ga = (unsigned long long)base;
  v4u g0;
  g0.x = 1u | (1u << 31);                       // count=1 | gather_mode, 16-bit idx
  g0.y = lds_off;
  g0.z = (unsigned)ga;
  g0.w = (unsigned)((ga >> 32) & 0x01FFFFFFu) | (2u << 30);
  v8i_t g1;
  g1[0] = 0x00010000;                           // data_size=1 (2 bytes)
  g1[1] = (int)((tensor_d0 & 0xFFFFu) << 16);
  g1[2] = (int)((tensor_d0 >> 16) | ((nrows & 0xFFFFu) << 16));
  g1[3] = (int)((nrows >> 16) | (tensor_d0 << 16));  // tile_dim0 = full row
  g1[4] = nvalid & 0xFFFF;                      // tile_dim1 = #indices
  g1[5] = (int)tensor_d0;                       // row stride
  g1[6] = 0;
  g1[7] = 0;
  v4i_t g2, g3;
#pragma unroll
  for (int q = 0; q < 4; ++q)
    g2[q] = (idx[2 * q] & 0xFFFF) | ((idx[2 * q + 1] & 0xFFFF) << 16);
#pragma unroll
  for (int q = 0; q < 4; ++q)
    g3[q] = (idx[8 + 2 * q] & 0xFFFF) | ((idx[8 + 2 * q + 1] & 0xFFFF) << 16);
  tdm_issue(g0, g1, g2, g3);
}
#else
#define USE_TDM 0
#endif

// ---------------- Phase 0: weight transpose + fp32->bf16 ----------------
// dst[c][r] = bf16(src[r][c]) per expert (blockIdx.z), 32x32 LDS tiles.
__global__ __launch_bounds__(256) void transpose_convert_kernel(
    const float* __restrict__ src, unsigned short* __restrict__ dst,
    int R, int C) {
  __shared__ float tile[32][33];
  const float* s = src + (size_t)blockIdx.z * R * C;
  unsigned short* d = dst + (size_t)blockIdx.z * R * C;
  int r0 = blockIdx.y * 32, c0 = blockIdx.x * 32;
  int j = threadIdx.x & 31, i0 = threadIdx.x >> 5;
#pragma unroll
  for (int rr = 0; rr < 4; ++rr) {
    int i = i0 + rr * 8;
    tile[i][j] = s[(size_t)(r0 + i) * C + c0 + j];
  }
  __syncthreads();
#pragma unroll
  for (int rr = 0; rr < 4; ++rr) {
    int i = i0 + rr * 8;
    d[(size_t)(c0 + i) * R + r0 + j] = f2bf(tile[j][i]);
  }
}

// ---------------- Phase 1: gating, routing lists, x -> bf16 ----------------
__global__ __launch_bounds__(256) void moe_gate_kernel(
    const float* __restrict__ x, const float* __restrict__ gate_w,
    int* __restrict__ counts, float* __restrict__ probsum,
    float* __restrict__ psel, int* __restrict__ tok_list,
    unsigned short* __restrict__ xb) {
  int wave = threadIdx.x >> 5;
  int lane = threadIdx.x & 31;
  int t = blockIdx.x * 8 + wave;
  if (t >= T_TOK) return;

  float acc[NEXP];
#pragma unroll
  for (int e = 0; e < NEXP; ++e) acc[e] = 0.f;

  const float* xr = x + (size_t)t * DMODEL;
  unsigned short* xbr = xb + (size_t)t * DMODEL;
#pragma unroll 4
  for (int j = lane; j < DMODEL; j += 32) {
    float xv = xr[j];
    xbr[j] = f2bf(xv);
    const float4* gw = (const float4*)(gate_w + (size_t)j * NEXP);
    float4 g0 = gw[0], g1 = gw[1];
    acc[0] += xv * g0.x; acc[1] += xv * g0.y;
    acc[2] += xv * g0.z; acc[3] += xv * g0.w;
    acc[4] += xv * g1.x; acc[5] += xv * g1.y;
    acc[6] += xv * g1.z; acc[7] += xv * g1.w;
  }
#pragma unroll
  for (int off = 16; off >= 1; off >>= 1)
#pragma unroll
    for (int e = 0; e < NEXP; ++e)
      acc[e] += __shfl_xor(acc[e], off, 32);

  float mx = acc[0];
#pragma unroll
  for (int e = 1; e < NEXP; ++e) mx = fmaxf(mx, acc[e]);
  float prob[NEXP], s = 0.f;
#pragma unroll
  for (int e = 0; e < NEXP; ++e) { prob[e] = __expf(acc[e] - mx); s += prob[e]; }
  float inv = 1.f / s;
  int gate = 0; float best = prob[0];
#pragma unroll
  for (int e = 1; e < NEXP; ++e)
    if (prob[e] > best) { best = prob[e]; gate = e; }  // first-max tie-break

  if (lane == 0) {
    int slot = atomicAdd(&counts[gate], 1);
    tok_list[gate * T_TOK + slot] = t;
    psel[t] = best * inv;
#pragma unroll
    for (int e = 0; e < NEXP; ++e) atomicAdd(&probsum[e], prob[e] * inv);
  }
}

// ---------------- Phase 2: expert FFN (WMMA + TDM pipeline) ----------------
// Grid (128 tiles, 8 experts), 512 threads = 16 waves.
// LDS: x[64x768] | h[64x64] | W1chunk[64 h][768 k] | W2chunk[768 c][64 k]
//      all bf16 -> 296 KB of the 320 KB WGP LDS.
// Pipeline: W2(ch) DMA overlaps phase A; W1(ch+1) DMA overlaps phase B.
extern __shared__ char smem_raw[];

__global__ __launch_bounds__(NTHREADS) void moe_ffn_kernel(
    const unsigned short* __restrict__ xb,
    const unsigned short* __restrict__ W1T,  // [e][h][k] bf16
    const unsigned short* __restrict__ W2T,  // [e][c][k] bf16
    const float* __restrict__ b1, const float* __restrict__ b2,
    const int* __restrict__ counts, const int* __restrict__ tok_list,
    const float* __restrict__ psel, float* __restrict__ out) {
  const int e = blockIdx.y;
  const int cnt = counts[e];
  const int m0 = blockIdx.x * MT;
  if (m0 >= cnt) return;
  const int nt = min(MT, cnt - m0);

  unsigned short* lds_x  = (unsigned short*)smem_raw;   // 64*768
  unsigned short* lds_h  = lds_x + MT * DMODEL;         // 64*64
  unsigned short* lds_w1 = lds_h + MT * HC;             // 64*768  [h][k]
  unsigned short* lds_w2 = lds_w1 + HC * DMODEL;        // 768*64  [c][k]

  const int tid = threadIdx.x, lane = tid & 31, w = tid >> 5;
  const int* tl = tok_list + e * T_TOK + m0;
  const unsigned short* W1e = W1T + (size_t)e * HDIM * DMODEL;
  const unsigned short* W2e = W2T + (size_t)e * DMODEL * HDIM;

#if USE_TDM
  if (w == 0) {
    // gather 64 token rows of xb via 4 gather-mode TDM descriptors
#pragma unroll
    for (int d = 0; d < 4; ++d) {
      int idx[16];
#pragma unroll
      for (int i = 0; i < 16; ++i) {
        int m = d * 16 + i;
        idx[i] = (m < nt) ? tl[m] : tl[0];
      }
      int nv = nt - d * 16;
      nv = nv < 0 ? 0 : (nv > 16 ? 16 : nv);
      if (nv > 0)
        tdm_gather_rows((unsigned)(unsigned long long)(lds_x + d * 16 * DMODEL),
                        xb, idx, nv, DMODEL, T_TOK);
    }
    // W1 chunk 0: 64 contiguous rows of 768 bf16
    tdm_load_2d((unsigned)(unsigned long long)lds_w1, W1e,
                DMODEL, HC, DMODEL, HC, DMODEL);
    // W2 chunk 0: 768 rows x 64 bf16, row stride HDIM
    tdm_load_2d((unsigned)(unsigned long long)lds_w2, W2e,
                HDIM, DMODEL, HC, DMODEL, HDIM);
    __builtin_amdgcn_s_wait_tensorcnt(1);  // x + W1(0) done; W2(0) in flight
  }
  __syncthreads();
#else
  // fallback: synchronous staging of x
  {
    int m = tid >> 3, part = tid & 7;
    int tok = (m < nt) ? tl[m] : -1;
    const unsigned short* sx = xb + (size_t)(tok >= 0 ? tok : 0) * DMODEL + part * 96;
    unsigned short* dx = lds_x + m * DMODEL + part * 96;
#pragma unroll 4
    for (int it = 0; it < 12; ++it) {
      uint4 v = make_uint4(0u, 0u, 0u, 0u);
      if (tok >= 0) v = *(const uint4*)(sx + it * 8);
      *(uint4*)(dx + it * 8) = v;
    }
  }
#endif

  const int mA = w & 3, nA = (w >> 2) & 3;  // phase-A tile (4x4 over 64x64)
  const int mB = w & 3, ng = (w >> 2) & 3;  // phase-B: N-tiles ng + 4j

  v8f acc[12];
#pragma unroll
  for (int j = 0; j < 12; ++j)
    acc[j] = (v8f){0.f, 0.f, 0.f, 0.f, 0.f, 0.f, 0.f, 0.f};

  for (int ch = 0; ch < NCH; ++ch) {
#if !USE_TDM
    {  // stage W1 chunk (contiguous bf16 copy)
      const unsigned short* sw = W1e + (size_t)ch * HC * DMODEL;
#pragma unroll 2
      for (int idx = tid; idx < HC * DMODEL / 8; idx += NTHREADS)
        *(uint4*)(lds_w1 + idx * 8) = *(const uint4*)(sw + idx * 8);
    }
    __syncthreads();
#endif

    // ---- phase A: h(16x16 per wave) = gelu(x @ W1chunk + b1) ----
    {
      v8f h = (v8f){0.f, 0.f, 0.f, 0.f, 0.f, 0.f, 0.f, 0.f};
#pragma unroll 4
      for (int ks = 0; ks < DMODEL / 32; ++ks) {
        v16bf a = load_frag_A(lds_x, mA * 16, ks * 32, DMODEL, lane);
        v16bf b = load_frag_B(lds_w1, nA * 16, ks * 32, DMODEL, lane);
        h = __builtin_amdgcn_wmma_f32_16x16x32_bf16(false, a, false, b,
                                                    (short)0, h, false, false);
      }
      int hcol = ch * HC + nA * 16 + (lane & 15);
      float b1v = b1[e * HDIM + hcol];
      unsigned short* hp =
          lds_h + (mA * 16 + (lane >> 4) * 8) * HC + nA * 16 + (lane & 15);
#pragma unroll
      for (int r = 0; r < 8; ++r) {
        float v = h[r] + b1v;
        v = 0.5f * v * (1.f + erff(v * 0.70710678118f));  // exact gelu
        hp[r * HC] = f2bf(v);
      }
    }
    __syncthreads();

#if USE_TDM
    if (w == 0) {
      __builtin_amdgcn_s_wait_tensorcnt(0);  // W2(ch) arrived
      if (ch + 1 < NCH)                      // W1(ch+1) overlaps phase B
        tdm_load_2d((unsigned)(unsigned long long)lds_w1,
                    W1e + (size_t)(ch + 1) * HC * DMODEL,
                    DMODEL, HC, DMODEL, HC, DMODEL);
    }
    __syncthreads();
#else
    {  // stage W2 chunk: 768 rows, 64 bf16 each (row stride HDIM)
      const unsigned short* sw = W2e + (size_t)ch * HC;
#pragma unroll 2
      for (int idx = tid; idx < DMODEL * (HC / 8); idx += NTHREADS) {
        int c = idx >> 3, q = idx & 7;
        *(uint4*)(lds_w2 + c * HC + q * 8) =
            *(const uint4*)(sw + (size_t)c * HDIM + q * 8);
      }
    }
    __syncthreads();
#endif

    // ---- phase B: out(64x768) += h(64x64) @ W2chunk(64x768) ----
    {
      v16bf a0 = load_frag_A(lds_h, mB * 16, 0,  HC, lane);
      v16bf a1 = load_frag_A(lds_h, mB * 16, 32, HC, lane);
#pragma unroll
      for (int j = 0; j < 12; ++j) {
        int n = ng + 4 * j;
        v16bf bb0 = load_frag_B(lds_w2, n * 16, 0,  HC, lane);
        acc[j] = __builtin_amdgcn_wmma_f32_16x16x32_bf16(
            false, a0, false, bb0, (short)0, acc[j], false, false);
        v16bf bb1 = load_frag_B(lds_w2, n * 16, 32, HC, lane);
        acc[j] = __builtin_amdgcn_wmma_f32_16x16x32_bf16(
            false, a1, false, bb1, (short)0, acc[j], false, false);
      }
    }
    __syncthreads();

#if USE_TDM
    if (w == 0 && ch + 1 < NCH) {
      __builtin_amdgcn_s_wait_tensorcnt(0);  // W1(ch+1) landed before phase A
      tdm_load_2d((unsigned)(unsigned long long)lds_w2,  // W2(ch+1) overlaps A
                  W2e + (size_t)(ch + 1) * HC,
                  HDIM, DMODEL, HC, DMODEL, HDIM);
    }
    __syncthreads();
#endif
  }

  // ---- epilogue: + b2, * p_sel, scatter to routed token rows ----
  int c_lo = lane & 15, g = lane >> 4;
  int   tokr[8];
  float pr[8];
#pragma unroll
  for (int r = 0; r < 8; ++r) {
    int m = mB * 16 + g * 8 + r;
    bool valid = m < nt;
    tokr[r] = valid ? tl[m] : -1;
    pr[r] = valid ? psel[tokr[r]] : 0.f;
  }
#pragma unroll
  for (int j = 0; j < 12; ++j) {
    int col = (ng + 4 * j) * 16 + c_lo;
    float b2v = b2[e * DMODEL + col];
#pragma unroll
    for (int r = 0; r < 8; ++r) {
      if (tokr[r] >= 0)
        out[(size_t)tokr[r] * DMODEL + col] = (acc[j][r] + b2v) * pr[r];
    }
  }
}

// ---------------- Phase 3: balance loss + counts ----------------
__global__ void moe_finalize_kernel(const int* __restrict__ counts,
                                    const float* __restrict__ probsum,
                                    float* __restrict__ tail) {
  if (threadIdx.x == 0 && blockIdx.x == 0) {
    float loss = 0.f;
    for (int e = 0; e < NEXP; ++e)
      loss += (probsum[e] / (float)T_TOK) * ((float)counts[e] / (float)T_TOK);
    tail[0] = loss * (float)NEXP;
    for (int e = 0; e < NEXP; ++e) tail[1 + e] = (float)counts[e];
  }
}

extern "C" void kernel_launch(void* const* d_in, const int* in_sizes, int n_in,
                              void* d_out, int out_size, void* d_ws, size_t ws_size,
                              hipStream_t stream) {
  const float* x      = (const float*)d_in[0];
  // d_in[1] attention_mask: all-ones, unused by reference math
  const float* gate_w = (const float*)d_in[2];
  const float* W1     = (const float*)d_in[3];
  const float* b1     = (const float*)d_in[4];
  const float* W2     = (const float*)d_in[5];
  const float* b2     = (const float*)d_in[6];
  float* out = (float*)d_out;

  // workspace layout (~88.4 MB): xb | W1T | W2T | counts | probsum | psel | lists
  const size_t XB_BYTES = (size_t)T_TOK * DMODEL * 2;
  const size_t WT_BYTES = (size_t)NEXP * DMODEL * HDIM * 2;
  char* wsb = (char*)d_ws;
  unsigned short* xb  = (unsigned short*)wsb;
  unsigned short* W1T = (unsigned short*)(wsb + XB_BYTES);
  unsigned short* W2T = (unsigned short*)(wsb + XB_BYTES + WT_BYTES);
  char* tail = wsb + XB_BYTES + 2 * WT_BYTES;
  int*   counts   = (int*)tail;
  float* probsum  = (float*)(tail + 32);
  float* psel     = (float*)(tail + 64);
  int*   tok_list = (int*)(tail + 64 + T_TOK * 4);

  (void)hipMemsetAsync(tail, 0, 64, stream);  // zero counts+probsum (capture-safe)

  // W1T[e][h][k] = bf16(W1[e][k][h]);  W2T[e][c][k] = bf16(W2[e][k][c])
  transpose_convert_kernel<<<dim3(HDIM / 32, DMODEL / 32, NEXP), 256, 0,
                             stream>>>(W1, W1T, DMODEL, HDIM);
  transpose_convert_kernel<<<dim3(DMODEL / 32, HDIM / 32, NEXP), 256, 0,
                             stream>>>(W2, W2T, HDIM, DMODEL);

  moe_gate_kernel<<<T_TOK / 8, 256, 0, stream>>>(x, gate_w, counts, probsum,
                                                 psel, tok_list, xb);

  size_t lds_bytes =
      (size_t)(MT * DMODEL + MT * HC + HC * DMODEL + DMODEL * HC) *
      sizeof(unsigned short);  // 296 KB of the 320 KB WGP LDS
  dim3 grid(T_TOK / MT, NEXP);
  moe_ffn_kernel<<<grid, NTHREADS, lds_bytes, stream>>>(
      xb, W1T, W2T, b1, b2, counts, tok_list, psel, out);

  moe_finalize_kernel<<<1, 32, 0, stream>>>(counts, probsum,
                                            out + (size_t)T_TOK * DMODEL);
}
